// cpcPretrainModel_61761629717004
// MI455X (gfx1250) — matrix operand order repaired and checked by reading, not compiled, not run on hardware
//
#include <hip/hip_runtime.h>

#define BSZ   64
#define SEQL  512
#define DDIM  256
#define HDIM  256
#define G4H   1024
#define NSP   5
#define NFNEG 9
#define TSTEPS (SEQL - NSP)   // 507

#define GEMM_K   256          // all our GEMMs have K == 256
#define KPAD     264          // padded LDS row stride (elements) to spread banks

typedef __attribute__((ext_vector_type(16))) __bf16 v16bf;
typedef __attribute__((ext_vector_type(8)))  float  v8f;
typedef __attribute__((ext_vector_type(4)))  unsigned int u32x4;

union Frag {
  v16bf bf;
  u32x4 u[2];
};

__device__ __forceinline__ float sigmoid_f(float x) {
  return 1.0f / (1.0f + __expf(-x));
}
__device__ __forceinline__ float tanh_f(float x) {
  return 2.0f / (1.0f + __expf(-2.0f * x)) - 1.0f;
}

// ---------------------------------------------------------------------------
// fp32 -> bf16 conversion
// ---------------------------------------------------------------------------
__global__ void f32_to_bf16_kernel(const float* __restrict__ in,
                                   __bf16* __restrict__ out, int n) {
  int i = blockIdx.x * blockDim.x + threadIdx.x;
  if (i < n) out[i] = (__bf16)in[i];
}

// ---------------------------------------------------------------------------
// WMMA GEMM: C[m,n] = sum_k A[m,k] * W[n,k] (+ bias0[n] + bias1[n])
// Block = 8 waves = 128 M-rows x 64 N-cols. The 64-row B strip of W (32 KB
// bf16) is shared by all 8 waves: staged once into LDS with
// global_load_async_to_lds_b128 (ASYNCcnt), then consumed via ds_load_b128.
// A fragments stream straight from global (rows are private per wave).
// amode==0: A row m at A + m*K
// amode==1: A row m=(s*64+b) at A + (b*SEQL + s)*K  (the [B,S,D] input "h")
// ---------------------------------------------------------------------------
__global__ __launch_bounds__(256)
void wmma_gemm_bf16(const __bf16* __restrict__ A,
                    const __bf16* __restrict__ W,
                    const float* __restrict__ bias0,
                    const float* __restrict__ bias1,
                    float* __restrict__ C,
                    int M, int N, int K, int amode) {
  __shared__ __align__(16) __bf16 smB[64 * KPAD];   // 33 KB

  const int tid  = threadIdx.x;
  const int wv   = tid >> 5;        // 0..7 (M sub-tile)
  const int lane = tid & 31;
  const int ln   = lane & 15;
  const int lh   = lane >> 4;

  const int tilesN = N >> 6;
  const int bm = blockIdx.x / tilesN;      // 128-row block index
  const int tn = blockIdx.x % tilesN;
  const int n0 = tn << 6;

  // ---- async-stage the B strip (rows n0..n0+63 of W, contiguous) into LDS
  {
    const char* src = (const char*)(W + (long)n0 * K);
    const unsigned dstbase = (unsigned)(uintptr_t)(&smB[0]);
    const int cpr = K >> 3;                 // 16B chunks per row (32)
    for (int u = tid; u < 64 * cpr; u += 256) {
      const int row = u / cpr;
      const int ch  = u % cpr;
      unsigned    l = dstbase + (unsigned)(row * (KPAD * 2) + ch * 16);
      const char* g = src + row * (K * 2) + ch * 16;
      asm volatile("global_load_async_to_lds_b128 %0, %1, off"
                   :: "v"(l), "v"(g) : "memory");
    }
    asm volatile("s_wait_asynccnt 0x0" ::: "memory");
  }
  __syncthreads();

  const int tm = bm * 8 + wv;
  const int m  = tm * 16 + ln;
  const long arow = (amode == 0)
                        ? (long)m * K
                        : ((long)(m & (BSZ - 1)) * SEQL + (m >> 6)) * (long)K;
  const __bf16* aptr = A + arow;

  const v8f vzero = {0.f, 0.f, 0.f, 0.f, 0.f, 0.f, 0.f, 0.f};
  v8f acc[4];
#pragma unroll
  for (int j = 0; j < 4; ++j) acc[j] = vzero;

  for (int k0 = 0; k0 < K; k0 += 32) {
    Frag a;
    a.u[0] = *(const u32x4*)(aptr + k0 + lh * 16);
    a.u[1] = *(const u32x4*)(aptr + k0 + lh * 16 + 8);
    if (k0 + 32 < K) __builtin_prefetch(aptr + k0 + 32, 0, 3);

    Frag b[4];
#pragma unroll
    for (int j = 0; j < 4; ++j) {
      const __bf16* bls = &smB[(j * 16 + ln) * KPAD + k0 + lh * 16];
      b[j].u[0] = *(const u32x4*)(bls);
      b[j].u[1] = *(const u32x4*)(bls + 8);
    }
#pragma unroll
    for (int j = 0; j < 4; ++j) {
      acc[j] = __builtin_amdgcn_wmma_f32_16x16x32_bf16(
          false, a.bf, false, b[j].bf, (short)0, acc[j], false, false);
    }
  }

#pragma unroll
  for (int j = 0; j < 4; ++j) {
    const int n = n0 + j * 16 + ln;
    float bj = 0.f;
    if (bias0) bj += bias0[n];
    if (bias1) bj += bias1[n];
#pragma unroll
    for (int r = 0; r < 8; ++r) {
      const int mm = tm * 16 + r + lh * 8;
      C[(long)mm * N + n] = acc[j][r] + bj;
    }
  }
}

// ---------------------------------------------------------------------------
// Persistent single-WGP LSTM scan over 512 steps.
// 512 threads = 16 waves. Wave w owns H-columns [w*16, w*16+16) of all 4
// gates and all 4 batch tiles. c stays in registers; h double-buffered in LDS
// (bf16, A-operand source). Gates GEMM [64,256]x[256,1024] via WMMA bf16;
// Whh stays L2-resident (512 KB bf16, reused 512x).
// ---------------------------------------------------------------------------
__global__ __launch_bounds__(512)
void lstm_layer_kernel(const float* __restrict__ xg,       // [512][64][1024]
                       const __bf16* __restrict__ WhhB,    // [1024][256]
                       __bf16* __restrict__ outB) {        // [512][64][256]
  __shared__ __align__(16) __bf16 hbuf[2][BSZ * HDIM];     // 2 x 32 KB

  const int tid  = threadIdx.x;
  const int wave = tid >> 5;   // 0..15
  const int lane = tid & 31;
  const int ln   = lane & 15;
  const int lh   = lane >> 4;

  // h_0 = 0
  for (int i = tid; i < BSZ * HDIM; i += 512) hbuf[0][i] = (__bf16)0.0f;
  __syncthreads();

  float c[4][8];
#pragma unroll
  for (int mt = 0; mt < 4; ++mt)
#pragma unroll
    for (int r = 0; r < 8; ++r) c[mt][r] = 0.f;

  // loop-invariant B row pointers (one Whh row per gate for this lane)
  const __bf16* brow[4];
#pragma unroll
  for (int g = 0; g < 4; ++g)
    brow[g] = WhhB + (long)(g * 256 + wave * 16 + ln) * HDIM + lh * 16;

  const v8f vzero = {0.f, 0.f, 0.f, 0.f, 0.f, 0.f, 0.f, 0.f};
  int p = 0;

  for (int t = 0; t < SEQL; ++t) {
    const float* xgt = xg + (size_t)t * (BSZ * G4H);

    for (int mt = 0; mt < 4; ++mt) {
      v8f acc[4];
#pragma unroll
      for (int g = 0; g < 4; ++g) acc[g] = vzero;

      const __bf16* arow = &hbuf[p][(mt * 16 + ln) * HDIM];
      for (int k0 = 0; k0 < HDIM; k0 += 32) {
        Frag a;
        a.u[0] = *(const u32x4*)(arow + k0 + lh * 16);
        a.u[1] = *(const u32x4*)(arow + k0 + lh * 16 + 8);
        Frag b[4];
#pragma unroll
        for (int g = 0; g < 4; ++g) {
          b[g].u[0] = *(const u32x4*)(brow[g] + k0);
          b[g].u[1] = *(const u32x4*)(brow[g] + k0 + 8);
        }
#pragma unroll
        for (int g = 0; g < 4; ++g) {
          acc[g] = __builtin_amdgcn_wmma_f32_16x16x32_bf16(
              false, a.bf, false, b[g].bf, (short)0, acc[g], false, false);
        }
      }

      // activations + state update for this batch tile
      const int kc = wave * 16 + ln;   // H column owned by this lane
#pragma unroll
      for (int r = 0; r < 8; ++r) {
        const int mm = mt * 16 + r + lh * 8;       // batch index
        const float* xr = xgt + mm * G4H + kc;
        float gi = sigmoid_f(acc[0][r] + xr[0]);
        float gf = sigmoid_f(acc[1][r] + xr[256]);
        float gg = tanh_f(acc[2][r] + xr[512]);
        float go = sigmoid_f(acc[3][r] + xr[768]);
        float cn = gf * c[mt][r] + gi * gg;
        c[mt][r] = cn;
        float hv = go * tanh_f(cn);
        __bf16 hb = (__bf16)hv;
        hbuf[p ^ 1][mm * HDIM + kc] = hb;
        outB[((size_t)t * BSZ + mm) * HDIM + kc] = hb;
      }
    }
    __syncthreads();
    p ^= 1;
  }
}

// ---------------------------------------------------------------------------
// CPC loss stage 1: one wave per (t,b); 10 candidate dot-products of length
// 256, wave32 shuffle reduction, logsumexp - preds[0] -> partial[t*64+b].
// ---------------------------------------------------------------------------
__global__ __launch_bounds__(256)
void cpc_partial_kernel(const float* __restrict__ q,     // [512][64][256]
                        const float* __restrict__ h,     // [64][512][256]
                        const int* __restrict__ idxs,    // [507][64*9]
                        float* __restrict__ partial) {   // [507*64]
  const int wid  = blockIdx.x * 8 + (threadIdx.x >> 5);
  const int lane = threadIdx.x & 31;
  if (wid >= TSTEPS * BSZ) return;
  const int t = wid >> 6;
  const int b = wid & 63;

  float qr[8];
  {
    const float* qq = q + (long)wid * DDIM + lane * 8;
#pragma unroll
    for (int d = 0; d < 8; ++d) qr[d] = qq[d];
  }

  float preds[1 + NFNEG];
#pragma unroll
  for (int n = 0; n < 1 + NFNEG; ++n) {
    const float* cp;
    if (n == 0) {
      cp = h + ((long)b * SEQL + (t + NSP)) * DDIM;          // positive
    } else {
      int idx  = idxs[t * (BSZ * NFNEG) + b * NFNEG + (n - 1)];
      int seqi = idx % (SEQL - 1);
      int bat  = idx / (SEQL - 1);
      if (seqi >= t + NSP) seqi += 1;                         // skip positive
      cp = h + ((long)bat * SEQL + seqi) * DDIM;
    }
    const float* cc = cp + lane * 8;
    float s = 0.f;
#pragma unroll
    for (int d = 0; d < 8; ++d) s += qr[d] * cc[d];
#pragma unroll
    for (int off = 16; off > 0; off >>= 1) s += __shfl_xor(s, off, 32);
    preds[n] = s;
  }

  float mx = preds[0];
#pragma unroll
  for (int n = 1; n < 1 + NFNEG; ++n) mx = fmaxf(mx, preds[n]);
  float se = 0.f;
#pragma unroll
  for (int n = 0; n < 1 + NFNEG; ++n) se += __expf(preds[n] - mx);
  float per = __logf(se) + mx - preds[0];
  if (lane == 0) partial[wid] = per;
}

// Deterministic final sum: loss = sum_{t,b} per / BSZ
__global__ __launch_bounds__(256)
void final_reduce_kernel(const float* __restrict__ partial,
                         float* __restrict__ out, int n) {
  __shared__ float sm[256];
  const int tid = threadIdx.x;
  float s = 0.f;
  for (int i = tid; i < n; i += 256) s += partial[i];
  sm[tid] = s;
  __syncthreads();
  for (int st = 128; st > 0; st >>= 1) {
    if (tid < st) sm[tid] += sm[tid + st];
    __syncthreads();
  }
  if (tid == 0) out[0] = sm[0] / (float)BSZ;
}

// ---------------------------------------------------------------------------
// Host-side orchestration
// ---------------------------------------------------------------------------
extern "C" void kernel_launch(void* const* d_in, const int* in_sizes, int n_in,
                              void* d_out, int out_size, void* d_ws, size_t ws_size,
                              hipStream_t stream) {
  (void)in_sizes; (void)n_in; (void)out_size; (void)ws_size;

  const float* h     = (const float*)d_in[0];
  const float* Wih0  = (const float*)d_in[1];
  const float* Whh0  = (const float*)d_in[2];
  const float* bih0  = (const float*)d_in[3];
  const float* bhh0  = (const float*)d_in[4];
  const float* Wih1  = (const float*)d_in[5];
  const float* Whh1  = (const float*)d_in[6];
  const float* bih1  = (const float*)d_in[7];
  const float* bhh1  = (const float*)d_in[8];
  const float* projw = (const float*)d_in[9];
  const int*   idxs  = (const int*)d_in[10];

  char* ws = (char*)d_ws;
  size_t off = 0;
  auto take = [&](size_t bytes) -> char* {
    char* p = ws + off;
    off += (bytes + 255) & ~(size_t)255;
    return p;
  };

  __bf16* hB    = (__bf16*)take((size_t)BSZ * SEQL * DDIM * 2);
  __bf16* Wih0B = (__bf16*)take((size_t)G4H * DDIM * 2);
  __bf16* Whh0B = (__bf16*)take((size_t)G4H * HDIM * 2);
  __bf16* Wih1B = (__bf16*)take((size_t)G4H * HDIM * 2);
  __bf16* Whh1B = (__bf16*)take((size_t)G4H * HDIM * 2);
  __bf16* PwB   = (__bf16*)take((size_t)DDIM * HDIM * 2);
  float*  xgbuf = (float*)take((size_t)SEQL * BSZ * G4H * 4);
  __bf16* out1B = (__bf16*)take((size_t)SEQL * BSZ * HDIM * 2);
  __bf16* out2B = (__bf16*)take((size_t)SEQL * BSZ * HDIM * 2);
  float*  oproj = (float*)take((size_t)SEQL * BSZ * DDIM * 4);
  float*  part  = (float*)take((size_t)TSTEPS * BSZ * 4);

  auto cvt = [&](const float* src, __bf16* dst, int n) {
    f32_to_bf16_kernel<<<(n + 255) / 256, 256, 0, stream>>>(src, dst, n);
  };
  cvt(h,     hB,    BSZ * SEQL * DDIM);
  cvt(Wih0,  Wih0B, G4H * DDIM);
  cvt(Whh0,  Whh0B, G4H * HDIM);
  cvt(Wih1,  Wih1B, G4H * HDIM);
  cvt(Whh1,  Whh1B, G4H * HDIM);
  cvt(projw, PwB,   DDIM * HDIM);

  const int M = SEQL * BSZ;                   // 32768
  // xg0 = h @ Wih0^T + bih0 + bhh0  (amode 1: A rows gathered from [B,S,D])
  {
    const int blocks = (M / 128) * (G4H / 64);   // 256*16 = 4096
    wmma_gemm_bf16<<<blocks, 256, 0, stream>>>(
        hB, Wih0B, bih0, bhh0, xgbuf, M, G4H, DDIM, 1);
  }
  lstm_layer_kernel<<<1, 512, 0, stream>>>(xgbuf, Whh0B, out1B);

  // xg1 = out1 @ Wih1^T + bih1 + bhh1
  {
    const int blocks = (M / 128) * (G4H / 64);
    wmma_gemm_bf16<<<blocks, 256, 0, stream>>>(
        out1B, Wih1B, bih1, bhh1, xgbuf, M, G4H, HDIM, 0);
  }
  lstm_layer_kernel<<<1, 512, 0, stream>>>(xgbuf, Whh1B, out2B);

  // output = out2 @ proj_w^T (no bias)
  {
    const int blocks = (M / 128) * (DDIM / 64);  // 256*4 = 1024
    wmma_gemm_bf16<<<blocks, 256, 0, stream>>>(
        out2B, PwB, nullptr, nullptr, oproj, M, DDIM, HDIM, 0);
  }

  cpc_partial_kernel<<<(TSTEPS * BSZ) / 8, 256, 0, stream>>>(
      oproj, h, idxs, part);
  final_reduce_kernel<<<1, 256, 0, stream>>>(part, (float*)d_out, TSTEPS * BSZ);
}
